// QGATLayer_v2_88905823027431
// MI455X (gfx1250) — compile-verified, matrix operand order checked by reading, not run on hardware
//
#include <hip/hip_runtime.h>

// ---------------------------------------------------------------------------
// Types for CDNA5 WMMA (wave32): bf16 operands, f32 accumulators.
// ---------------------------------------------------------------------------
typedef __attribute__((ext_vector_type(16))) __bf16 v16bf;
typedef __attribute__((ext_vector_type(8)))  __bf16 v8bf;
typedef __attribute__((ext_vector_type(4)))  __bf16 v4bf;
typedef __attribute__((ext_vector_type(8)))  float  v8f;
typedef __attribute__((ext_vector_type(4)))  int    v4i;

union bfvec { v16bf v; v8bf h[2]; };

#define NEGBIG (-9.0e15f)   // reference NEG_INF (finite, so exp() underflows cleanly)

// ---------------------------------------------------------------------------
// CDNA5 async global->LDS path (ASYNCcnt-tracked), guarded so the file also
// compiles if the toolchain lacks the builtin.
// Builtin prototype (from the round-2 diagnostic): (v4i AS1*, v4i AS3*, Ii, Ii).
// ---------------------------------------------------------------------------
#if defined(__AMDGCN__) && __has_builtin(__builtin_amdgcn_global_load_async_to_lds_b128)
#define HAVE_ASYNC_LDS 1
#else
#define HAVE_ASYNC_LDS 0
#endif

__device__ __forceinline__ void wait_async0() {
#if defined(__AMDGCN__)
#if __has_builtin(__builtin_amdgcn_s_wait_asynccnt)
  __builtin_amdgcn_s_wait_asynccnt(0);
#else
  asm volatile("s_wait_asynccnt 0x0" ::: "memory");
#endif
#endif
}

#if HAVE_ASYNC_LDS
typedef __attribute__((address_space(1))) v4i* gv4i_p;
typedef __attribute__((address_space(3))) v4i* lv4i_p;
__device__ __forceinline__ void async_b128_to_lds(const int* gptr, int* lptr) {
  __builtin_amdgcn_global_load_async_to_lds_b128((gv4i_p)gptr, (lv4i_p)lptr, 0, 0);
}
#endif

// ---------------------------------------------------------------------------
// Kernel 0: build hamilton^T in bf16.  hamT[q][p] = hamilton[p][q],
// hamilton = signed block permutation of W (256x256).  hamT: 256 x 1024.
// ---------------------------------------------------------------------------
__global__ __launch_bounds__(256) void build_hamT(const float* __restrict__ W,
                                                  __bf16* __restrict__ hamT) {
  int idx = blockIdx.x * 256 + threadIdx.x;   // idx = q*1024 + p, 262144 total
  int q = idx >> 10, p = idx & 1023;
  int bi = p >> 8, wr = p & 255;              // hamilton row block / W row
  int bj = q >> 6, cw = q & 63;               // hamilton col block / col-in-block
  const int   srct[4][4] = {{0,1,2,3},{1,0,3,2},{2,3,0,1},{3,2,1,0}};
  const float sgnt[4][4] = {{1.f,-1.f,-1.f,-1.f},
                            {1.f, 1.f,-1.f, 1.f},
                            {1.f, 1.f, 1.f,-1.f},
                            {1.f,-1.f, 1.f, 1.f}};
  float v = sgnt[bi][bj] * W[wr * 256 + srct[bi][bj] * 64 + cw];
  hamT[idx] = (__bf16)v;
}

// ---------------------------------------------------------------------------
// Kernel 1: input f32 -> bf16 (row-major 8192x1024), vectorized x4.
// ---------------------------------------------------------------------------
__global__ __launch_bounds__(256) void cvt_input(const float* __restrict__ in,
                                                 __bf16* __restrict__ outp) {
  int i = (blockIdx.x * 256 + threadIdx.x) * 4;
  float4 f = *(const float4*)(in + i);
  v4bf o;
  o[0] = (__bf16)f.x; o[1] = (__bf16)f.y; o[2] = (__bf16)f.z; o[3] = (__bf16)f.w;
  *(v4bf*)(outp + i) = o;
}

// ---------------------------------------------------------------------------
// Kernel 2: h = input @ hamilton  via V_WMMA_F32_16X16X32_BF16.
// One wave per 16x64 output strip; K-loop of 32 (1024/32), double-buffered
// in registers so next-iteration loads issue under current WMMAs (partial
// s_wait_loadcnt instead of full stalls).
// Output written TRANSPOSED in bf16: hT[n][m], 256 x 8192.
// ---------------------------------------------------------------------------
__global__ __launch_bounds__(32) void gemm1_kernel(const __bf16* __restrict__ Abf, // 8192x1024
                                                   const __bf16* __restrict__ BT,  // 256x1024 (hamT)
                                                   __bf16* __restrict__ hT) {      // 256x8192
  const int lane = threadIdx.x & 31;
  const int m0 = blockIdx.x * 16;
  const int nstrip = blockIdx.y * 64;
  const int ML = lane & 15;                 // A: row M, B: col N (within tile)
  const int kb = (lane >> 4) ? 8 : 0;
  v8f acc[4] = {};
  bfvec a[2];
  bfvec b[2][4];

  auto load_tiles = [&](int kt, int sbuf) {
    const int k0 = kt * 32;
    const __bf16* ap = Abf + (long)(m0 + ML) * 1024 + k0 + kb;
    a[sbuf].h[0] = *(const v8bf*)(ap);
    a[sbuf].h[1] = *(const v8bf*)(ap + 16);
#pragma unroll
    for (int nt = 0; nt < 4; ++nt) {
      const __bf16* bp = BT + (long)(nstrip + nt * 16 + ML) * 1024 + k0 + kb;
      b[sbuf][nt].h[0] = *(const v8bf*)(bp);
      b[sbuf][nt].h[1] = *(const v8bf*)(bp + 16);
    }
  };

  load_tiles(0, 0);
  for (int kt = 0; kt < 32; ++kt) {
    const int cur = kt & 1;
    if (kt + 1 < 32) load_tiles(kt + 1, cur ^ 1);   // prefetch next K-slab
#pragma unroll
    for (int nt = 0; nt < 4; ++nt) {
      acc[nt] = __builtin_amdgcn_wmma_f32_16x16x32_bf16(
          false, a[cur].v, false, b[cur][nt].v, (short)0, acc[nt], false, false);
    }
  }

  const int mo = (lane >> 4) ? 8 : 0;
#pragma unroll
  for (int nt = 0; nt < 4; ++nt) {
    const int N = nstrip + nt * 16 + ML;
#pragma unroll
    for (int v = 0; v < 8; ++v) {
      hT[(long)N * 8192 + m0 + v + mo] = (__bf16)acc[nt][v];
    }
  }
}

// ---------------------------------------------------------------------------
// Kernel 3: s = h @ a1, t = h @ a2 (GEMVs).  Reads hT column-wise =>
// coalesced across threads (thread = m).
// ---------------------------------------------------------------------------
__global__ __launch_bounds__(256) void st_kernel(const __bf16* __restrict__ hT,
                                                 const float* __restrict__ a,
                                                 float* __restrict__ s,
                                                 float* __restrict__ t) {
  int m = blockIdx.x * 256 + threadIdx.x;
  float sa = 0.f, ta = 0.f;
  for (int n = 0; n < 256; ++n) {
    float hv = (float)hT[(long)n * 8192 + m];
    sa += hv * a[n];
    ta += hv * a[256 + n];
  }
  s[m] = sa;
  t[m] = ta;
}

// ---------------------------------------------------------------------------
// Kernel 4: zero BN-stat accumulators.
// ---------------------------------------------------------------------------
__global__ void zero_stats(float* csum, float* csq) {
  int i = threadIdx.x;
  if (i < 256) { csum[i] = 0.f; csq[i] = 0.f; }
}

// ---------------------------------------------------------------------------
// Kernel 5: fused masked-softmax attention (flash style).
// Block = 256 thr (8 waves) owns 16 rows; streams 16x32 key tiles.
// adj tile comes in via double-buffered async global->LDS (ASYNCcnt path);
// hT B-tiles are issued at the top of the iteration so the whole softmax
// section hides their latency before the WMMAs consume them.
// ---------------------------------------------------------------------------
__global__ __launch_bounds__(256) void flash_kernel(const int* __restrict__ adj,
                                                    const float* __restrict__ s,
                                                    const float* __restrict__ t,
                                                    const __bf16* __restrict__ hT,
                                                    float* __restrict__ hp) {
  __shared__ __bf16 pA[16 * 32];
  __shared__ float mcur[16], lcur[16], scl[16], rmax[16];
#if HAVE_ASYNC_LDS
  __shared__ int adjT[2][16 * 32];
#endif

  const int tid  = threadIdx.x;
  const int lane = tid & 31;
  const int wave = tid >> 5;
  const int i0   = blockIdx.x * 16;
  const int r    = tid >> 4;            // row 0..15 (16 threads per row)
  const int c2   = (tid & 15) * 2;      // 2 columns per thread
  const int n0   = wave * 32;           // this wave's first N tile
  const int Nl   = lane & 15;
  const int kb   = (lane >> 4) ? 8 : 0;
  const int mo   = (lane >> 4) ? 8 : 0;

  if (tid < 16) { mcur[tid] = -3.0e38f; lcur[tid] = 0.f; }

  const float s_r = s[i0 + r];
  const long  arow = (long)(i0 + r) * 8192;
  v8f acc0 = {}, acc1 = {};

#if HAVE_ASYNC_LDS
  // 128 lanes each move one b128 (4 ints): 16 rows x 8 chunks.
  const int arr = tid >> 3;             // 0..15 (valid when tid < 128)
  const int acc4 = (tid & 7) * 4;       // 0,4,...,28
  if (tid < 128) {
    async_b128_to_lds(adj + (long)(i0 + arr) * 8192 + 0 + acc4, &adjT[0][arr * 32 + acc4]);
  }
#endif
  __syncthreads();

  for (int jt = 0; jt < 256; ++jt) {
    const int j0 = jt * 32;
    // ---- issue this iteration's B tiles early (consumed by WMMA at the end)
    bfvec b0, b1;
    {
      const __bf16* bp0 = hT + (long)(n0 + Nl) * 8192 + j0 + kb;
      const __bf16* bp1 = hT + (long)(n0 + 16 + Nl) * 8192 + j0 + kb;
      b0.h[0] = *(const v8bf*)(bp0);  b0.h[1] = *(const v8bf*)(bp0 + 16);
      b1.h[0] = *(const v8bf*)(bp1);  b1.h[1] = *(const v8bf*)(bp1 + 16);
    }
    // ---- adj tile ----
#if HAVE_ASYNC_LDS
    wait_async0();          // tile jt has landed in LDS
    __syncthreads();
    const int buf = jt & 1;
    int2 am = *(const int2*)(&adjT[buf][r * 32 + c2]);
    if (jt + 1 < 256 && tid < 128) {   // kick off tile jt+1 into the other buffer
      async_b128_to_lds(adj + (long)(i0 + arr) * 8192 + (j0 + 32) + acc4,
                        &adjT[buf ^ 1][arr * 32 + acc4]);
    }
#else
    int2 am;
    am.x = __builtin_nontemporal_load(adj + arow + j0 + c2);
    am.y = __builtin_nontemporal_load(adj + arow + j0 + c2 + 1);
    if (jt < 255)
      __builtin_prefetch(adj + arow + j0 + 32 + c2, 0, 0);   // global_prefetch_b8
#endif
    // ---- e tile: leaky_relu(s_i + t_j), masked by adj ----
    float e0 = s_r + t[j0 + c2];
    float e1 = s_r + t[j0 + c2 + 1];
    e0 = e0 > 0.f ? e0 : 0.2f * e0;
    e1 = e1 > 0.f ? e1 : 0.2f * e1;
    e0 = am.x > 0 ? e0 : NEGBIG;
    e1 = am.y > 0 ? e1 : NEGBIG;
    // ---- row max (16-lane xor reduce within wave half) ----
    float mt = fmaxf(e0, e1);
    for (int d = 1; d < 16; d <<= 1) mt = fmaxf(mt, __shfl_xor(mt, d, 16));
    if ((tid & 15) == 0) rmax[r] = mt;
    __syncthreads();
    // ---- online stats update ----
    if (tid < 16) {
      float mold = mcur[tid];
      float mnew = fmaxf(mold, rmax[tid]);
      float sc   = __expf(mold - mnew);
      mcur[tid]  = mnew;
      lcur[tid] *= sc;
      scl[tid]   = sc;
    }
    __syncthreads();
    // ---- probabilities -> LDS bf16 tile + row sums ----
    const float mr = mcur[r];
    float p0 = __expf(e0 - mr);
    float p1 = __expf(e1 - mr);
    pA[r * 32 + c2]     = (__bf16)p0;
    pA[r * 32 + c2 + 1] = (__bf16)p1;
    float ps = p0 + p1;
    for (int d = 1; d < 16; d <<= 1) ps += __shfl_xor(ps, d, 16);
    if ((tid & 15) == 0) lcur[r] += ps;
    __syncthreads();
    // ---- rescale accumulators, then WMMA: acc += P_tile @ h_tile ----
#pragma unroll
    for (int v = 0; v < 8; ++v) {
      float scv = scl[v + mo];
      acc0[v] *= scv;
      acc1[v] *= scv;
    }
    bfvec a;
    a.h[0] = *(const v8bf*)(&pA[(lane & 15) * 32 + kb]);
    a.h[1] = *(const v8bf*)(&pA[(lane & 15) * 32 + kb + 16]);
    acc0 = __builtin_amdgcn_wmma_f32_16x16x32_bf16(
        false, a.v, false, b0.v, (short)0, acc0, false, false);
    acc1 = __builtin_amdgcn_wmma_f32_16x16x32_bf16(
        false, a.v, false, b1.v, (short)0, acc1, false, false);
    __syncthreads();
  }
  // ---- finalize: divide by softmax denominator, store h_prime (f32) ----
#pragma unroll
  for (int v = 0; v < 8; ++v) {
    const int M = v + mo;
    const float inv = 1.0f / lcur[M];
    hp[(long)(i0 + M) * 256 + n0 + Nl]      = acc0[v] * inv;
    hp[(long)(i0 + M) * 256 + n0 + 16 + Nl] = acc1[v] * inv;
  }
}

// ---------------------------------------------------------------------------
// Kernel 6: per-column sum / sumsq partials (block = 32 rows, thread = col).
// ---------------------------------------------------------------------------
__global__ __launch_bounds__(256) void bn_stats(const float* __restrict__ hp,
                                                float* __restrict__ csum,
                                                float* __restrict__ csq) {
  const int c = threadIdx.x;
  const int r0 = blockIdx.x * 32;
  float su = 0.f, q = 0.f;
  for (int r = 0; r < 32; ++r) {
    float v = hp[(long)(r0 + r) * 256 + c];
    su += v;
    q  += v * v;
  }
  atomicAdd(&csum[c], su);
  atomicAdd(&csq[c], q);
}

// ---------------------------------------------------------------------------
// Kernel 7: batch-norm + ELU, final f32 output (8192 x 256).
// ---------------------------------------------------------------------------
__global__ __launch_bounds__(256) void bn_final(const float* __restrict__ hp,
                                                const float* __restrict__ csum,
                                                const float* __restrict__ csq,
                                                const float* __restrict__ gamma,
                                                const float* __restrict__ beta,
                                                float* __restrict__ out) {
  const int c = threadIdx.x;
  const long i = blockIdx.x;
  const float invN = 1.0f / 8192.0f;
  float mean = csum[c] * invN;
  float var  = csq[c] * invN - mean * mean;
  float x = (hp[i * 256 + c] - mean) * rsqrtf(var + 1e-5f) * gamma[c] + beta[c];
  out[i * 256 + c] = x > 0.f ? x : (__expf(x) - 1.0f);
}

// ---------------------------------------------------------------------------
// Host launcher.
// ---------------------------------------------------------------------------
extern "C" void kernel_launch(void* const* d_in, const int* in_sizes, int n_in,
                              void* d_out, int out_size, void* d_ws, size_t ws_size,
                              hipStream_t stream) {
  const float* input = (const float*)d_in[0];   // 8192 x 1024
  const int*   adj   = (const int*)d_in[1];     // 8192 x 8192
  const float* W     = (const float*)d_in[2];   // 256 x 256
  const float* a     = (const float*)d_in[3];   // 512 x 1
  const float* gamma = (const float*)d_in[4];   // 256
  const float* beta  = (const float*)d_in[5];   // 256
  float* out = (float*)d_out;                   // 8192 x 256 f32

  char* ws = (char*)d_ws;
  size_t off = 0;
  __bf16* inBf = (__bf16*)(ws + off); off += (size_t)8192 * 1024 * 2; // 16 MB
  __bf16* hamT = (__bf16*)(ws + off); off += (size_t)256 * 1024 * 2;  // 512 KB
  __bf16* hT   = (__bf16*)(ws + off); off += (size_t)256 * 8192 * 2;  // 4 MB
  float*  sv   = (float*)(ws + off);  off += (size_t)8192 * 4;
  float*  tv   = (float*)(ws + off);  off += (size_t)8192 * 4;
  float*  hp   = (float*)(ws + off);  off += (size_t)8192 * 256 * 4;  // 8 MB
  float*  csum = (float*)(ws + off);  off += 256 * 4;
  float*  csq  = (float*)(ws + off);  off += 256 * 4;

  build_hamT<<<1024, 256, 0, stream>>>(W, hamT);
  cvt_input<<<8192, 256, 0, stream>>>(input, inBf);
  gemm1_kernel<<<dim3(512, 4), 32, 0, stream>>>(inBf, hamT, hT);
  st_kernel<<<32, 256, 0, stream>>>(hT, a, sv, tv);
  zero_stats<<<1, 256, 0, stream>>>(csum, csq);
  flash_kernel<<<512, 256, 0, stream>>>(adj, sv, tv, hT, hp);
  bn_stats<<<256, 256, 0, stream>>>(hp, csum, csq);
  bn_final<<<8192, 256, 0, stream>>>(hp, csum, csq, gamma, beta, out);
}
